// DINOBevAligner_56315611185831
// MI455X (gfx1250) — compile-verified, matrix operand order checked by reading, not run on hardware
//
#include <hip/hip_runtime.h>
#include <hip/hip_bf16.h>

// ---- geometry constants ----
#define BEV_H   80
#define BEV_W   80
#define QN      (BEV_H * BEV_W)      // 6400
#define DPIL    4
#define VNUM    6
#define HP      37
#define WPGRID  37
#define CDIM    768
#define NTOK    (HP * WPGRID)        // 1369
#define SCALE_F 0.32375f             // 518/1600
#define W2M1    517.0f
#define H2M1    517.0f

typedef __attribute__((ext_vector_type(16))) __bf16 v16bf;
typedef __attribute__((ext_vector_type(8)))  float  v8f;

union FragU { unsigned u[8]; v16bf v; };

// ---------------------------------------------------------------------------
// Kernel 0: convert proj_w (768x768 f32, row-major [c_out][c_in]) -> bf16
// ---------------------------------------------------------------------------
__global__ __launch_bounds__(256) void k_cvt_w(const float* __restrict__ w,
                                               __bf16* __restrict__ wb, int n) {
  int i = blockIdx.x * 256 + threadIdx.x;
  if (i < n) wb[i] = (__bf16)w[i];
}

// ---------------------------------------------------------------------------
// Kernel 1: fused projection + bilinear sampling + LayerNorm + masked depth
// fusion + softplus view fusion.  One wave32 per BEV query; lane L owns
// channels L, L+32, ..., L+736 (24 channels) -> coalesced 128B lines.
// Output: f_bev (Q x C) in bf16 (feeds the WMMA GEMM).
// ---------------------------------------------------------------------------
__global__ __launch_bounds__(256) void k_sample_fuse(
    const float* __restrict__ tokens,   // (V, NTOK, C)
    const float* __restrict__ l2i,      // (V, 4, 4)
    const float* __restrict__ gamma,    // (C)
    const float* __restrict__ beta,     // (C)
    const float* __restrict__ wview,    // (V)
    __bf16* __restrict__ fbev)          // (Q, C) bf16
{
  const int lane = threadIdx.x & 31;
  const int wv   = threadIdx.x >> 5;
  const int q    = blockIdx.x * 8 + wv;
  if (q >= QN) return;

  const int   qy = q / BEV_W, qx = q - qy * BEV_W;
  const float px = ((qx + 0.5f) / (float)BEV_W) * 102.4f - 51.2f;
  const float py = ((qy + 0.5f) / (float)BEV_H) * 102.4f - 51.2f;

  float g[24], bt[24];
#pragma unroll
  for (int j = 0; j < 24; ++j) { int c = lane + 32 * j; g[j] = gamma[c]; bt[j] = beta[c]; }

  float acc[24];
#pragma unroll
  for (int j = 0; j < 24; ++j) acc[j] = 0.0f;
  float denAll = 0.0f;

  for (int v = 0; v < VNUM; ++v) {
    const float* M  = l2i + v * 16;
    const float m00 = M[0],  m01 = M[1],  m02 = M[2],  m03 = M[3];
    const float m10 = M[4],  m11 = M[5],  m12 = M[6],  m13 = M[7];
    const float m20 = M[8],  m21 = M[9],  m22 = M[10], m23 = M[11];
    const float* fm = tokens + (size_t)v * NTOK * CDIM;

    float accD[24];
#pragma unroll
    for (int j = 0; j < 24; ++j) accD[j] = 0.0f;
    float mcnt = 0.0f;

    for (int d = 0; d < DPIL; ++d) {
      const float pz = (0.5f + d * (7.0f / 3.0f)) - 5.0f;
      const float cx = m00 * px + m01 * py + m02 * pz + m03;
      const float cy = m10 * px + m11 * py + m12 * pz + m13;
      const float cz = m20 * px + m21 * py + m22 * pz + m23;
      const float invd = 1.0f / fmaxf(cz, 1e-5f);
      const float ud = cx * invd * SCALE_F;
      const float vd = cy * invd * SCALE_F;
      const bool ok = (cz > 1e-5f) & (ud >= 0.0f) & (ud <= W2M1) & (vd >= 0.0f) & (vd <= H2M1);
      if (!ok) continue;
      mcnt += 1.0f;

      // grid_sample(border, align_corners=False) coords on the 37x37 patch grid
      const float gx = 2.0f * (ud * (1.0f / 14.0f)) * (1.0f / W2M1) - 1.0f;
      const float gy = 2.0f * (vd * (1.0f / 14.0f)) * (1.0f / H2M1) - 1.0f;
      float x = fminf(fmaxf(((gx + 1.0f) * (float)WPGRID - 1.0f) * 0.5f, 0.0f), (float)(WPGRID - 1));
      float y = fminf(fmaxf(((gy + 1.0f) * (float)HP     - 1.0f) * 0.5f, 0.0f), (float)(HP - 1));
      const float x0f = floorf(x), y0f = floorf(y);
      const float wx = x - x0f,   wy = y - y0f;
      const int x0 = (int)x0f, y0 = (int)y0f;
      const int x1 = min(x0 + 1, WPGRID - 1), y1 = min(y0 + 1, HP - 1);
      const float* r00 = fm + ((size_t)(y0 * WPGRID + x0)) * CDIM;
      const float* r01 = fm + ((size_t)(y0 * WPGRID + x1)) * CDIM;
      const float* r10 = fm + ((size_t)(y1 * WPGRID + x0)) * CDIM;
      const float* r11 = fm + ((size_t)(y1 * WPGRID + x1)) * CDIM;
      const float w00 = (1.0f - wx) * (1.0f - wy), w01 = wx * (1.0f - wy);
      const float w10 = (1.0f - wx) * wy,          w11 = wx * wy;

      float f[24];
      float psum = 0.0f;
#pragma unroll
      for (int j = 0; j < 24; ++j) {
        const int c = lane + 32 * j;
        const float s = w00 * r00[c] + w01 * r01[c] + w10 * r10[c] + w11 * r11[c];
        f[j] = s; psum += s;
      }
      // wave32 reduction for LN mean
#pragma unroll
      for (int off = 16; off > 0; off >>= 1) psum += __shfl_xor(psum, off, 32);
      const float mu = psum * (1.0f / (float)CDIM);
      float pvar = 0.0f;
#pragma unroll
      for (int j = 0; j < 24; ++j) { const float dd = f[j] - mu; pvar += dd * dd; }
#pragma unroll
      for (int off = 16; off > 0; off >>= 1) pvar += __shfl_xor(pvar, off, 32);
      const float rstd = rsqrtf(pvar * (1.0f / (float)CDIM) + 1e-5f);
#pragma unroll
      for (int j = 0; j < 24; ++j) accD[j] += (f[j] - mu) * rstd * g[j] + bt[j];
    }

    const float denom = fmaxf(mcnt, 1e-6f);
    const float xw = wview[v];
    const float sw = fmaxf(xw, 0.0f) + log1pf(expf(-fabsf(xw)));  // softplus
    const float vval = (denom > 0.0f) ? 1.0f : 0.0f;
    const float inv = 1.0f / denom;
#pragma unroll
    for (int j = 0; j < 24; ++j) acc[j] += accD[j] * inv * sw;
    denAll += sw * vval;
  }

  const float invD = 1.0f / fmaxf(denAll, 1e-6f);
#pragma unroll
  for (int j = 0; j < 24; ++j)
    fbev[(size_t)q * CDIM + lane + 32 * j] = (__bf16)(acc[j] * invD);
}

// ---------------------------------------------------------------------------
// Kernel 2: WMMA GEMM  out[co][q] = sum_ci fbev[q][ci] * W[co][ci] + b[co]
// v_wmma_f32_16x16x32_bf16. Each wave computes a 16(M=q) x 64(N=co) strip:
// A fragment loaded once per K-step and reused by 4 WMMAs.
// Documented 16-bit A 16x32 layout: lane<16 -> K {0..7,16..23} pairs,
// lane>=16 -> K {8..15,24..31}; B symmetric with lane owning column N.
// ---------------------------------------------------------------------------
__global__ __launch_bounds__(256) void k_wmma_proj(
    const __bf16* __restrict__ fbev,    // (Q, C) bf16
    const __bf16* __restrict__ wb,      // (C_out, C_in) bf16
    const float*  __restrict__ bias,    // (C_out)
    float* __restrict__ out)            // (C_out, Q)
{
  const int lane = threadIdx.x & 31;
  const int wv   = threadIdx.x >> 5;
  const int gid  = blockIdx.x * 8 + wv;      // global wave id
  const int numM = QN / 16;                  // 400
  const int tm   = gid % numM;               // M tile (16 queries)
  const int tn   = gid / numM;               // N strip (64 out channels)
  if (tn >= CDIM / 64) return;

  const int l15  = lane & 15;
  const int hiK  = (lane < 16) ? 0 : 8;      // pair-index shift for upper half-wave

  // packed-pair (dword) offsets within a 32-wide K tile, per VGPR slot j
  int koff[8];
#pragma unroll
  for (int j = 0; j < 8; ++j)
    koff[j] = ((j < 4) ? (2 * j) : (16 + 2 * (j - 4))) + hiK;   // even element idx

  const unsigned* arow = (const unsigned*)(fbev + (size_t)(tm * 16 + l15) * CDIM);
  const unsigned* brow[4];
#pragma unroll
  for (int s = 0; s < 4; ++s)
    brow[s] = (const unsigned*)(wb + (size_t)(tn * 64 + s * 16 + l15) * CDIM);

  v8f accv[4];
#pragma unroll
  for (int s = 0; s < 4; ++s) accv[s] = (v8f){0.f,0.f,0.f,0.f,0.f,0.f,0.f,0.f};

  for (int kb = 0; kb < CDIM; kb += 32) {
    FragU aF;
#pragma unroll
    for (int j = 0; j < 8; ++j) aF.u[j] = arow[(kb + koff[j]) >> 1];
#pragma unroll
    for (int s = 0; s < 4; ++s) {
      FragU bF;
#pragma unroll
      for (int j = 0; j < 8; ++j) bF.u[j] = brow[s][(kb + koff[j]) >> 1];
      accv[s] = __builtin_amdgcn_wmma_f32_16x16x32_bf16(
          /*neg_a=*/false, aF.v, /*neg_b=*/false, bF.v,
          /*c_mod=*/(short)0, accv[s], /*reuse_a=*/false, /*reuse_b=*/false);
    }
  }

  // Epilogue: out is (C_out, Q) row-major => out[co*QN + q], plus bias.
  const int qbase = tm * 16 + ((lane < 16) ? 0 : 8);
#pragma unroll
  for (int s = 0; s < 4; ++s) {
    const int co = tn * 64 + s * 16 + l15;
    const float bco = bias[co];
    float* orow = out + (size_t)co * QN;
#pragma unroll
    for (int r = 0; r < 8; ++r)
      orow[qbase + r] = accv[s][r] + bco;
  }
}

// ---------------------------------------------------------------------------
extern "C" void kernel_launch(void* const* d_in, const int* in_sizes, int n_in,
                              void* d_out, int out_size, void* d_ws, size_t ws_size,
                              hipStream_t stream) {
  const float* tokens = (const float*)d_in[0];   // (1,6,1369,768)
  const float* l2i    = (const float*)d_in[1];   // (1,6,4,4)
  const float* gamma  = (const float*)d_in[2];   // (768)
  const float* beta   = (const float*)d_in[3];   // (768)
  const float* wview  = (const float*)d_in[4];   // (1,6,1)
  const float* projw  = (const float*)d_in[5];   // (768,768)
  const float* projb  = (const float*)d_in[6];   // (768)
  float* out = (float*)d_out;                    // (1,768,80,80)

  char* ws = (char*)d_ws;
  __bf16* wb   = (__bf16*)ws;                            // 768*768*2   = 1,179,648 B
  __bf16* fbev = (__bf16*)(ws + (size_t)CDIM * CDIM * 2);// 6400*768*2  = 9,830,400 B

  // K0: proj_w -> bf16
  k_cvt_w<<<(CDIM * CDIM + 255) / 256, 256, 0, stream>>>(projw, wb, CDIM * CDIM);
  // K1: fused sampling (one wave per query; 8 waves / 256-thread block)
  k_sample_fuse<<<QN / 8, 256, 0, stream>>>(tokens, l2i, gamma, beta, wview, fbev);
  // K2: WMMA projection (400 M-tiles x 12 N-strips = 4800 waves -> 600 blocks)
  k_wmma_proj<<<(QN / 16) * (CDIM / 64) / 8, 256, 0, stream>>>(fbev, wb, projb, out);
}
